// MultiHeadAttention_23373212025502
// MI455X (gfx1250) — matrix operand order repaired and checked by reading, not compiled, not run on hardware
//
#include <hip/hip_runtime.h>

typedef _Float16 f16;
typedef _Float16 v8h  __attribute__((ext_vector_type(8)));
typedef _Float16 v16h __attribute__((ext_vector_type(16)));
typedef float    v8f  __attribute__((ext_vector_type(8)));

#define D_MODEL 1024
#define N_HEADS 16
#define D_K     64
#define BATCH   4
#define SEQ     2048
#define M_TOT   (BATCH * SEQ)

__device__ __forceinline__ v8h ld8(const f16* p) { return *(const v8h*)p; }
__device__ __forceinline__ void st8(f16* p, v8h v) { *(v8h*)p = v; }

__device__ __forceinline__ v16h mk16(v8h a, v8h b) {
  v16h r;
#pragma unroll
  for (int i = 0; i < 8; i++) { r[i] = a[i]; r[i + 8] = b[i]; }
  return r;
}

__device__ __forceinline__ v8f wmma_f16(v16h a, v16h b, v8f c) {
  return __builtin_amdgcn_wmma_f32_16x16x32_f16(false, a, false, b, (short)0, c,
                                                false, false);
}

// ---- CDNA5 async global->LDS copy (ASYNCcnt-tracked, bypasses VGPRs) ------
// Per-lane: LDS[vdst] = MEM[vaddr], 16 bytes. GV mode (SADDR = off).
// LDS byte address = low 32 bits of the generic LDS pointer.
__device__ __forceinline__ void cp16_async(f16* lds_dst, const f16* gsrc) {
  unsigned lds_off = (unsigned)(size_t)lds_dst;
  asm volatile("global_load_async_to_lds_b128 %0, %1, off"
               :: "v"(lds_off), "v"(gsrc)
               : "memory");
}

__device__ __forceinline__ void async_fence() {
  asm volatile("s_wait_asynccnt 0x0" ::: "memory");
}

// ---------------------------------------------------------------------------
// Projection GEMM:  Y[M,1024] = X[M,1024] @ W[1024,1024]^T + bias
//   XF16     : X is f16 (attention output) vs f32 (raw inputs)
//   OUT_MODE : 0 = f32 flat [M,1024] (final output)
//              1 = f16 [B,H,S,64]   (Q, K)
//              2 = f16 [B,H,64,S]   (V, pre-transposed for attention B-frags)
// Block: 256 threads = 8 waves; block tile 128(M) x 64(N); K-chunk 64.
// Wave wv owns rows wv*16..wv*16+15 and all four 16-col N subtiles.
// ---------------------------------------------------------------------------
template <bool XF16, int OUT_MODE>
__global__ __launch_bounds__(256) void gemm_proj(
    const void* __restrict__ Xv, const float* __restrict__ W,
    const float* __restrict__ bias, void* __restrict__ Yv) {
  __shared__ f16 As[128][72];  // +8 pad, rows stay 16B aligned (144B stride)
  __shared__ f16 Bs[64][72];

  const int t = threadIdx.x;
  const int lane = t & 31, wv = t >> 5;
  const int half = lane >> 4, ln = lane & 15;
  const int m0 = blockIdx.y * 128, n0 = blockIdx.x * 64;

  const int srowA = t >> 1, scolA = (t & 1) * 32;  // A staging: 32 elems/thread
  const int srowB = t >> 2, scolB = (t & 3) * 16;  // B staging: 16 elems/thread

  v8f acc[4] = {};

  for (int kc = 0; kc < 1024; kc += 64) {
    // ---- stage A tile ----
    if (XF16) {
      const f16* xp = (const f16*)Xv + (size_t)(m0 + srowA) * 1024 + kc + scolA;
#pragma unroll
      for (int j = 0; j < 4; j++) cp16_async(&As[srowA][scolA + 8 * j], xp + 8 * j);
    } else {
      const float* xp = (const float*)Xv + (size_t)(m0 + srowA) * 1024 + kc + scolA;
#pragma unroll
      for (int j = 0; j < 4; j++) {
        v8h h;
#pragma unroll
        for (int i = 0; i < 8; i++) h[i] = (f16)xp[8 * j + i];
        st8(&As[srowA][scolA + 8 * j], h);
      }
      if (kc + 64 < 1024) __builtin_prefetch(xp + 64, 0, 0);
    }
    // ---- stage B tile: row n of Bs = row (n0+n) of W (== B-matrix column) ----
    {
      const float* wp = W + (size_t)(n0 + srowB) * 1024 + kc + scolB;
      v8h h0, h1;
#pragma unroll
      for (int i = 0; i < 8; i++) { h0[i] = (f16)wp[i]; h1[i] = (f16)wp[8 + i]; }
      st8(&Bs[srowB][scolB], h0);
      st8(&Bs[srowB][scolB + 8], h1);
      if (kc + 64 < 1024) __builtin_prefetch(wp + 64, 0, 0);
    }
    if (XF16) async_fence();
    __syncthreads();

#pragma unroll
    for (int k0 = 0; k0 < 64; k0 += 32) {
      // A fragment: lane row = wv*16+ln; K halves split per lane-half
      const int ar = wv * 16 + ln;
      v16h a = mk16(ld8(&As[ar][k0 + half * 8]),
                    ld8(&As[ar][k0 + 16 + half * 8]));
#pragma unroll
      for (int nt = 0; nt < 4; nt++) {
        v16h b = mk16(ld8(&Bs[nt * 16 + ln][k0 + half * 8]),
                      ld8(&Bs[nt * 16 + ln][k0 + 16 + half * 8]));
        acc[nt] = wmma_f16(a, b, acc[nt]);
      }
    }
    __syncthreads();
  }

  // ---- epilogue: bias + store ----
#pragma unroll
  for (int nt = 0; nt < 4; nt++) {
    const int col = n0 + nt * 16 + ln;
    const float bv = bias[col];
#pragma unroll
    for (int r = 0; r < 8; r++) {
      const int m = m0 + wv * 16 + r + 8 * half;  // C layout row
      const float v = acc[nt][r] + bv;
      if (OUT_MODE == 0) {         // f32 flat [M,1024]
        ((float*)Yv)[(size_t)m * D_MODEL + col] = v;
      } else {
        const int b = m >> 11, sq = m & 2047;
        const int h = col >> 6, d = col & 63;
        if (OUT_MODE == 1)         // f16 [B,H,S,64]
          ((f16*)Yv)[((((size_t)b * N_HEADS + h) * SEQ) + sq) * D_K + d] = (f16)v;
        else                       // f16 [B,H,64,S]  (transposed V)
          ((f16*)Yv)[((((size_t)b * N_HEADS + h) * D_K) + d) * SEQ + sq] = (f16)v;
      }
    }
  }
}

// ---------------------------------------------------------------------------
// Flash attention: one block per (b, h, 128-query tile); each wave owns 16 q.
// Online softmax over 64-key blocks; scores and PV both via WMMA.
// K is [B,H,S,64]; V is pre-transposed [B,H,64,S] so both stages are straight
// copies -> async global->LDS.
// ---------------------------------------------------------------------------
__global__ __launch_bounds__(256) void attn_kernel(
    const f16* __restrict__ Qh, const f16* __restrict__ Kh,
    const f16* __restrict__ Vtg, const float* __restrict__ mask,
    f16* __restrict__ AO) {
  __shared__ f16 Ks[64][64];    // K block, row-major [key][d]
  __shared__ f16 Vt[64][64];    // V block [d][key]
  __shared__ f16 Ps[128][64];   // per-wave P staging (wave w -> rows w*16..)

  const int t = threadIdx.x;
  const int lane = t & 31, wv = t >> 5;
  const int half = lane >> 4, ln = lane & 15;

  const int bx = blockIdx.x;
  const int qb = bx & 15, h = (bx >> 4) & 15, b = bx >> 8;
  const int q0 = qb * 128 + wv * 16;                      // wave's query base
  const size_t bhS = ((size_t)b * N_HEADS + h) * SEQ;     // K/Q row base
  const size_t bhD = ((size_t)b * N_HEADS + h) * D_K;     // Vt row base

  // Q fragments (K-chunks 0..31, 32..63); fold 1/sqrt(d_k)=0.125 into Q
  const f16* qp = Qh + (bhS + q0 + ln) * D_K;
  const v16h aq0 = mk16(ld8(qp + half * 8),      ld8(qp + 16 + half * 8)) * (f16)0.125f;
  const v16h aq1 = mk16(ld8(qp + 32 + half * 8), ld8(qp + 48 + half * 8)) * (f16)0.125f;

  float mrow[8], lrow[8];
  v8f oacc[4] = {};
#pragma unroll
  for (int r = 0; r < 8; r++) { mrow[r] = -1e30f; lrow[r] = 0.f; }

  const int srow = t >> 2, scol = (t & 3) * 16;

  for (int kb = 0; kb < SEQ; kb += 64) {
    // ---- stage K rows and transposed-V rows (pure copies -> async) ----
    {
      const f16* kp = Kh + (bhS + kb + srow) * D_K + scol;
      cp16_async(&Ks[srow][scol], kp);
      cp16_async(&Ks[srow][scol + 8], kp + 8);
      const f16* vp = Vtg + (bhD + srow) * SEQ + kb + scol;
      cp16_async(&Vt[srow][scol], vp);
      cp16_async(&Vt[srow][scol + 8], vp + 8);
    }
    async_fence();
    __syncthreads();

    // ---- scores S = (Q*0.125) @ K^T + mask_bias : 16 x 64 ----
    v8f sc[4];
#pragma unroll
    for (int kt = 0; kt < 4; kt++) {
      v8f s = {};
#pragma unroll
      for (int k0 = 0; k0 < 64; k0 += 32) {
        v16h bk = mk16(ld8(&Ks[kt * 16 + ln][k0 + half * 8]),
                       ld8(&Ks[kt * 16 + ln][k0 + 16 + half * 8]));
        s = wmma_f16((k0 == 0) ? aq0 : aq1, bk, s);
      }
      const float mb =
          (mask[(size_t)b * SEQ + kb + kt * 16 + ln] > 0.5f) ? 0.f : -1e9f;
#pragma unroll
      for (int r = 0; r < 8; r++) s[r] += mb;
      sc[kt] = s;
    }

    // ---- online softmax (rows in VGPR idx + lane-half; 16-lane shfl) ----
#pragma unroll
    for (int r = 0; r < 8; r++) {
      float mx = sc[0][r];
#pragma unroll
      for (int kt = 1; kt < 4; kt++) mx = fmaxf(mx, sc[kt][r]);
#pragma unroll
      for (int off = 1; off < 16; off <<= 1) mx = fmaxf(mx, __shfl_xor(mx, off, 32));
      const float mnew = fmaxf(mrow[r], mx);
      const float scl = __expf(mrow[r] - mnew);
      float ps = 0.f;
#pragma unroll
      for (int kt = 0; kt < 4; kt++) {
        const float p = __expf(sc[kt][r] - mnew);
        sc[kt][r] = p;
        ps += p;
      }
#pragma unroll
      for (int off = 1; off < 16; off <<= 1) ps += __shfl_xor(ps, off, 32);
      lrow[r] = lrow[r] * scl + ps;
      mrow[r] = mnew;
#pragma unroll
      for (int n = 0; n < 4; n++) oacc[n][r] *= scl;
    }

    // ---- P (C layout) -> per-wave LDS -> A layout; O += P @ V ----
#pragma unroll
    for (int kt = 0; kt < 4; kt++)
#pragma unroll
      for (int r = 0; r < 8; r++)
        Ps[wv * 16 + r + 8 * half][kt * 16 + ln] = (f16)sc[kt][r];

#pragma unroll
    for (int k0 = 0; k0 < 64; k0 += 32) {
      v16h ap = mk16(ld8(&Ps[wv * 16 + ln][k0 + half * 8]),
                     ld8(&Ps[wv * 16 + ln][k0 + 16 + half * 8]));
#pragma unroll
      for (int n = 0; n < 4; n++) {
        v16h bv = mk16(ld8(&Vt[n * 16 + ln][k0 + half * 8]),
                       ld8(&Vt[n * 16 + ln][k0 + 16 + half * 8]));
        oacc[n] = wmma_f16(ap, bv, oacc[n]);
      }
    }
    __syncthreads();
  }

  // ---- normalize, write f16 [B,S,1024] so output proj is a plain GEMM ----
#pragma unroll
  for (int r = 0; r < 8; r++) {
    const float inv = 1.f / lrow[r];
    const int q = q0 + r + 8 * half;
#pragma unroll
    for (int n = 0; n < 4; n++) {
      AO[((size_t)b * SEQ + q) * D_MODEL + h * 64 + n * 16 + ln] =
          (f16)(oacc[n][r] * inv);
    }
  }
}

// ---------------------------------------------------------------------------
extern "C" void kernel_launch(void* const* d_in, const int* in_sizes, int n_in,
                              void* d_out, int out_size, void* d_ws,
                              size_t ws_size, hipStream_t stream) {
  const float* query = (const float*)d_in[0];
  const float* key   = (const float*)d_in[1];
  const float* value = (const float*)d_in[2];
  const float* mask  = (const float*)d_in[3];
  const float* w_q = (const float*)d_in[4];
  const float* b_q = (const float*)d_in[5];
  const float* w_k = (const float*)d_in[6];
  const float* b_k = (const float*)d_in[7];
  const float* w_v = (const float*)d_in[8];
  const float* b_v = (const float*)d_in[9];
  const float* w_o = (const float*)d_in[10];
  const float* b_o = (const float*)d_in[11];

  const size_t elems = (size_t)BATCH * SEQ * D_MODEL;  // 8.39M
  f16* Qh  = (f16*)d_ws;       // [B,H,S,64]  f16 (16 MB)
  f16* Kh  = Qh + elems;       // [B,H,S,64]  f16 (16 MB)
  f16* Vtg = Kh + elems;       // [B,H,64,S]  f16 (16 MB)
  f16* AO  = Vtg + elems;      // [B,S,1024]  f16 (16 MB)

  dim3 gGemm(D_MODEL / 64, M_TOT / 128);  // 16 x 64 blocks

  gemm_proj<false, 1><<<gGemm, 256, 0, stream>>>(query, w_q, b_q, Qh);
  gemm_proj<false, 1><<<gGemm, 256, 0, stream>>>(key,   w_k, b_k, Kh);
  gemm_proj<false, 2><<<gGemm, 256, 0, stream>>>(value, w_v, b_v, Vtg);

  attn_kernel<<<BATCH * N_HEADS * (SEQ / 128), 256, 0, stream>>>(Qh, Kh, Vtg,
                                                                 mask, AO);

  gemm_proj<true, 0><<<gGemm, 256, 0, stream>>>(AO, w_o, b_o, d_out);
}
